// MultiHeadAttention_76553497084436
// MI455X (gfx1250) — matrix operand order repaired and checked by reading, not compiled, not run on hardware
//
#include <hip/hip_runtime.h>
#include <hip/hip_bf16.h>
#include <math.h>

typedef __attribute__((ext_vector_type(16))) __bf16 v16bf;
typedef __attribute__((ext_vector_type(8)))  __bf16 v8bf;
typedef __attribute__((ext_vector_type(8)))  float  v8f;

#define TDIM   2048
#define DMODEL 1024
#define NHEADS 16
#define HDIM   64
#define WIN    128

static __device__ inline v8f wmma_bf16(v16bf a, v16bf b, v8f c) {
  // D = A(16x32 bf16) * B(32x16 bf16) + C(16x16 f32)
  return __builtin_amdgcn_wmma_f32_16x16x32_bf16(false, a, false, b, (short)0, c,
                                                 false, false);
}

// A-fragment (16x32 bf16): lane row = lane%16, kbase = (lane/16)*16,
// halfs[0..15] = A[row][kbase+0..15].
static __device__ inline v16bf load_a_frag_f32(const float* p) {
  v16bf a;
#pragma unroll
  for (int i = 0; i < 8; ++i) a[i] = (__bf16)p[i];
#pragma unroll
  for (int i = 0; i < 8; ++i) a[8 + i] = (__bf16)p[8 + i];
  return a;
}
static __device__ inline v16bf load_a_frag_bf16(const __bf16* p) {
  v8bf lo = *(const v8bf*)p;
  v8bf hi = *(const v8bf*)(p + 8);
  return __builtin_shufflevector(lo, hi, 0, 1, 2, 3, 4, 5, 6, 7,
                                 8, 9, 10, 11, 12, 13, 14, 15);
}
// B-fragment (32x16 bf16): lane col = lane%16, kbase = (lane/16)*16,
// halfs[m] = B[kbase+m][col] -> contiguous when B stored column-major [col][K].
static __device__ inline v16bf load_b_frag(const __bf16* p) {
  return *(const v16bf*)p;
}

// ---------------- weight convert + transpose: W[K][N] f32 -> WT[N][K] bf16
__global__ void wtrans_kernel(const float* __restrict__ W,
                              __bf16* __restrict__ WT, int K, int Ncols) {
  int idx = blockIdx.x * blockDim.x + threadIdx.x;   // over K*Ncols, coalesced read
  int n = idx % Ncols;
  int k = idx / Ncols;
  WT[(size_t)n * K + k] = (__bf16)W[idx];
}

// ---------------- QKV GEMM (8192 x 3072) + scatter to q/k/vT (bf16)
// Block tile 128x256, 8 waves as 2x4, wave tile 64x64 (16 WMMA / k-step).
__global__ __launch_bounds__(256)
void qkv_gemm_kernel(const float* __restrict__ x, const __bf16* __restrict__ WqkvT,
                     __bf16* __restrict__ qbuf, __bf16* __restrict__ kbuf,
                     __bf16* __restrict__ vT) {
  const int lane = threadIdx.x & 31;
  const int w    = threadIdx.x >> 5;
  const int wr = w >> 2, wc = w & 3;
  const int hi = lane >> 4, ln = lane & 15;
  const int rowBase = blockIdx.y * 128 + wr * 64;
  const int colBase = blockIdx.x * 256 + wc * 64;

  v8f zero = {};
  v8f acc[4][4];
#pragma unroll
  for (int i = 0; i < 4; ++i)
#pragma unroll
    for (int j = 0; j < 4; ++j) acc[i][j] = zero;

  for (int kb = 0; kb < DMODEL; kb += 32) {
    v16bf afr[4];
#pragma unroll
    for (int rt = 0; rt < 4; ++rt) {
      int row = rowBase + rt * 16 + ln;
      afr[rt] = load_a_frag_f32(x + (size_t)row * DMODEL + kb + hi * 16);
    }
    v16bf bfr[4];
#pragma unroll
    for (int ct = 0; ct < 4; ++ct) {
      int col = colBase + ct * 16 + ln;
      bfr[ct] = load_b_frag(WqkvT + (size_t)col * DMODEL + kb + hi * 16);
    }
#pragma unroll
    for (int rt = 0; rt < 4; ++rt)
#pragma unroll
      for (int ct = 0; ct < 4; ++ct)
        acc[rt][ct] = wmma_bf16(afr[rt], bfr[ct], acc[rt][ct]);
  }

#pragma unroll
  for (int rt = 0; rt < 4; ++rt) {
#pragma unroll
    for (int ct = 0; ct < 4; ++ct) {
      int col = colBase + ct * 16 + ln;
      int s = col >> 10;
      int rem = col & 1023;
      int h = rem >> 6;
      int d = rem & 63;
#pragma unroll
      for (int r = 0; r < 8; ++r) {
        int g = rowBase + rt * 16 + r + hi * 8;   // global row = n*T + t
        int n = g >> 11, t = g & 2047;
        __bf16 v = (__bf16)acc[rt][ct][r];
        size_t nh = (size_t)(n * NHEADS + h);
        if (s == 0)      qbuf[(nh * TDIM + t) * HDIM + d] = v;
        else if (s == 1) kbuf[(nh * TDIM + t) * HDIM + d] = v;
        else             vT  [(nh * HDIM + d) * TDIM + t] = v;
      }
    }
  }
}

// ---------------- RoPE in place on [n*h][t][64], pairs (d, d+32)
__global__ void rope_kernel(__bf16* __restrict__ buf) {
  int idx = blockIdx.x * blockDim.x + threadIdx.x;  // N*H*T*32 threads
  int d  = idx & 31;
  int t  = (idx >> 5) & 2047;
  int nh = idx >> 16;
  size_t base = ((size_t)nh * TDIM + t) * HDIM;
  // inv_freq = ROPE_BASE^(-d/32); log(10000) = 9.210340371976184
  float inv = expf(-(float)d * (9.210340371976184f / 32.0f));
  float ang = (float)t * inv;
  float c = cosf(ang), s = sinf(ang);
  float x1 = (float)buf[base + d];
  float x2 = (float)buf[base + d + 32];
  buf[base + d]      = (__bf16)(x1 * c - x2 * s);
  buf[base + d + 32] = (__bf16)(x1 * s + x2 * c);
}

// ---------------- sliding-window flash attention, 1 wave = 16 query rows
__global__ __launch_bounds__(256)
void attn_kernel(const __bf16* __restrict__ qbuf, const __bf16* __restrict__ kbuf,
                 const __bf16* __restrict__ vT, __bf16* __restrict__ attn) {
  __shared__ __bf16 pls[8][16][32];                  // per-wave P staging (8 KB)
  const int lane = threadIdx.x & 31;
  const int w    = threadIdx.x >> 5;
  const int hi = lane >> 4, ln = lane & 15;
  const int nh = blockIdx.z * NHEADS + blockIdx.y;
  const int q0 = blockIdx.x * 128 + w * 16;

  const __bf16* qh = qbuf + (size_t)nh * TDIM * HDIM;
  const __bf16* kh = kbuf + (size_t)nh * TDIM * HDIM;
  const __bf16* vh = vT   + (size_t)nh * HDIM * TDIM;

  v16bf qa[2];
#pragma unroll
  for (int f = 0; f < 2; ++f)
    qa[f] = load_a_frag_bf16(qh + (size_t)(q0 + ln) * HDIM + f * 32 + hi * 16);

  v8f zero = {};
  v8f acc[4];
#pragma unroll
  for (int i = 0; i < 4; ++i) acc[i] = zero;
  float m[8], l[8];
#pragma unroll
  for (int r = 0; r < 8; ++r) { m[r] = -3.0e38f; l[r] = 0.0f; }

  int kstart = q0 - WIN; if (kstart < 0) kstart = 0;
  kstart &= ~31;                                     // 32-aligned; keys never OOB
  int kend = q0 + 16 + WIN; if (kend > TDIM) kend = TDIM;

  for (int ks = kstart; ks < kend; ks += 32) {
    v8f sc[2];
#pragma unroll
    for (int kt = 0; kt < 2; ++kt) {
      int key = ks + kt * 16 + ln;
      const __bf16* kp = kh + (size_t)key * HDIM + hi * 16;
      v16bf b0 = *(const v16bf*)(kp);                // d = 0..31
      v16bf b1 = *(const v16bf*)(kp + 32);           // d = 32..63
      v8f s0 = zero;
      s0 = wmma_bf16(qa[0], b0, s0);
      s0 = wmma_bf16(qa[1], b1, s0);
      sc[kt] = s0;
    }
    float pv[2][8];
#pragma unroll
    for (int r = 0; r < 8; ++r) {
      int row = q0 + r + hi * 8;
      float best = -3.0e38f;
#pragma unroll
      for (int kt = 0; kt < 2; ++kt) {
        int key = ks + kt * 16 + ln;
        int rel = key - row;
        bool keep = (rel >= -WIN) && (rel <= WIN);
        float sv = sc[kt][r] * 0.125f + (keep ? 0.0f : -3.0e38f);
        pv[kt][r] = sv;
        best = fmaxf(best, sv);
      }
      best = fmaxf(best, __shfl_xor(best, 1));
      best = fmaxf(best, __shfl_xor(best, 2));
      best = fmaxf(best, __shfl_xor(best, 4));
      best = fmaxf(best, __shfl_xor(best, 8));
      float mnew  = fmaxf(m[r], best);
      float alpha = __expf(m[r] - mnew);
      float rsum = 0.0f;
#pragma unroll
      for (int kt = 0; kt < 2; ++kt) {
        float p = __expf(pv[kt][r] - mnew);
        pv[kt][r] = p;
        rsum += p;
      }
      rsum += __shfl_xor(rsum, 1);
      rsum += __shfl_xor(rsum, 2);
      rsum += __shfl_xor(rsum, 4);
      rsum += __shfl_xor(rsum, 8);
      l[r] = l[r] * alpha + rsum;
      m[r] = mnew;
#pragma unroll
      for (int ct = 0; ct < 4; ++ct) acc[ct][r] = acc[ct][r] * alpha;
    }
    // stage P (C layout -> A layout) through wave-local LDS
#pragma unroll
    for (int kt = 0; kt < 2; ++kt)
#pragma unroll
      for (int r = 0; r < 8; ++r)
        pls[w][r + hi * 8][kt * 16 + ln] = (__bf16)pv[kt][r];
    asm volatile("s_wait_dscnt 0" ::: "memory");     // same-wave DS RAW fence
    v16bf pa = load_a_frag_bf16(&pls[w][ln][hi * 16]);
#pragma unroll
    for (int ct = 0; ct < 4; ++ct) {
      int d = ct * 16 + ln;
      v16bf bv = *(const v16bf*)(vh + (size_t)d * TDIM + ks + hi * 16);
      acc[ct] = wmma_bf16(pa, bv, acc[ct]);
    }
  }

#pragma unroll
  for (int ct = 0; ct < 4; ++ct)
#pragma unroll
    for (int r = 0; r < 8; ++r) {
      int t = q0 + r + hi * 8;
      float o = acc[ct][r] / l[r];
      attn[((size_t)blockIdx.z * TDIM + t) * DMODEL +
           blockIdx.y * HDIM + ct * 16 + ln] = (__bf16)o;
    }
}

// ---------------- output GEMM (8192 x 1024) + bias -> f32
// Block tile 128x256, 8 waves as 2x4, wave tile 64x64.
__global__ __launch_bounds__(256)
void out_gemm_kernel(const __bf16* __restrict__ attn, const __bf16* __restrict__ WoutT,
                     const float* __restrict__ bout, float* __restrict__ out) {
  const int lane = threadIdx.x & 31;
  const int w    = threadIdx.x >> 5;
  const int wr = w >> 2, wc = w & 3;
  const int hi = lane >> 4, ln = lane & 15;
  const int rowBase = blockIdx.y * 128 + wr * 64;
  const int colBase = blockIdx.x * 256 + wc * 64;

  v8f zero = {};
  v8f acc[4][4];
#pragma unroll
  for (int i = 0; i < 4; ++i)
#pragma unroll
    for (int j = 0; j < 4; ++j) acc[i][j] = zero;

  for (int kb = 0; kb < DMODEL; kb += 32) {
    v16bf afr[4];
#pragma unroll
    for (int rt = 0; rt < 4; ++rt) {
      int row = rowBase + rt * 16 + ln;
      afr[rt] = load_a_frag_bf16(attn + (size_t)row * DMODEL + kb + hi * 16);
    }
    v16bf bfr[4];
#pragma unroll
    for (int ct = 0; ct < 4; ++ct) {
      int col = colBase + ct * 16 + ln;
      bfr[ct] = load_b_frag(WoutT + (size_t)col * DMODEL + kb + hi * 16);
    }
#pragma unroll
    for (int rt = 0; rt < 4; ++rt)
#pragma unroll
      for (int ct = 0; ct < 4; ++ct)
        acc[rt][ct] = wmma_bf16(afr[rt], bfr[ct], acc[rt][ct]);
  }

#pragma unroll
  for (int rt = 0; rt < 4; ++rt)
#pragma unroll
    for (int ct = 0; ct < 4; ++ct) {
      int col = colBase + ct * 16 + ln;
      float b = bout[col];
#pragma unroll
      for (int r = 0; r < 8; ++r) {
        int g = rowBase + rt * 16 + r + hi * 8;
        out[(size_t)g * DMODEL + col] = acc[rt][ct][r] + b;
      }
    }
}

extern "C" void kernel_launch(void* const* d_in, const int* in_sizes, int n_in,
                              void* d_out, int out_size, void* d_ws, size_t ws_size,
                              hipStream_t stream) {
  const float* x    = (const float*)d_in[0];   // (4, 2048, 1024)
  const float* Wqkv = (const float*)d_in[1];   // (1024, 3072)
  const float* Wout = (const float*)d_in[2];   // (1024, 1024)
  const float* bout = (const float*)d_in[3];   // (1024,)
  float* out = (float*)d_out;

  char* ws = (char*)d_ws;
  __bf16* WqkvT = (__bf16*)(ws);                             //  6 MB: [3072][1024]
  __bf16* WoutT = (__bf16*)(ws + 6291456);                   //  2 MB: [1024][1024]
  __bf16* qbuf  = (__bf16*)(ws + 8388608);                   // 16 MB: [n*h][t][64]
  __bf16* kbuf  = (__bf16*)(ws + 8388608 + 16777216);        // 16 MB
  __bf16* vTb   = (__bf16*)(ws + 8388608 + 2 * 16777216);    // 16 MB: [n*h][64][t]
  __bf16* attn  = (__bf16*)(ws + 8388608 + 3 * 16777216);    // 16 MB: [n*t][1024]

  wtrans_kernel<<<12288, 256, 0, stream>>>(Wqkv, WqkvT, 1024, 3072);
  wtrans_kernel<<<4096, 256, 0, stream>>>(Wout, WoutT, 1024, 1024);
  qkv_gemm_kernel<<<dim3(12, 64), 256, 0, stream>>>(x, WqkvT, qbuf, kbuf, vTb);
  rope_kernel<<<16384, 256, 0, stream>>>(qbuf);
  rope_kernel<<<16384, 256, 0, stream>>>(kbuf);
  attn_kernel<<<dim3(16, NHEADS, 4), 256, 0, stream>>>(qbuf, kbuf, vTb, attn);
  out_gemm_kernel<<<dim3(4, 64), 256, 0, stream>>>(attn, WoutT, bout, out);
}
